// Linear_11424613007510
// MI455X (gfx1250) — matrix operand order repaired and checked by reading, not compiled, not run on hardware
//
#include <hip/hip_runtime.h>

// Problem dims (fixed by the reference)
#define B_DIM 2048
#define K_DIM 4096
#define N_DIM 4096

// Blocking
#define TILE_M 128
#define TILE_N 128
#define TILE_K 32
#define KP 40                 // padded K-stride in elements (80B, 16B-aligned rows)
#define NKC (K_DIM / TILE_K)  // 128 K-chunks

typedef __attribute__((ext_vector_type(16))) __bf16 v16bf;
typedef __attribute__((ext_vector_type(8)))  float  v8f;
typedef __attribute__((ext_vector_type(8)))  unsigned short v8us;
typedef __attribute__((ext_vector_type(4)))  unsigned short v4us;

union BfPack {
  v8us  h[2];
  v16bf bf;
};

__device__ __forceinline__ unsigned short bf_bits(__bf16 b) {
  union { __bf16 b; unsigned short u; } x;
  x.b = b;
  return x.u;
}

// Split f32 into bf16 hi/lo: hi = bf16_rne(f), lo = bf16_rne(f - hi).
// Native __bf16 casts let the compiler use the hardware bf16 converts.
__device__ __forceinline__ void bf_split(float f, unsigned short& hi, unsigned short& lo) {
  __bf16 h = (__bf16)f;
  __bf16 l = (__bf16)(f - (float)h);
  hi = bf_bits(h);
  lo = bf_bits(l);
}

// A fragment (16x32 bf16, MxK): lane l holds row M = l%16; elements 0..7 are
// K = half*8 + 0..7, elements 8..15 are K = 16 + half*8 + 0..7  (ISA 7.12.2).
__device__ __forceinline__ v16bf load_frag_a(const unsigned short* base, int row, int half) {
  BfPack p;
  p.h[0] = *reinterpret_cast<const v8us*>(base + row * KP + half * 8);
  p.h[1] = *reinterpret_cast<const v8us*>(base + row * KP + 16 + half * 8);
  return p.bf;
}
// B fragment (32x16 bf16, KxN): lane l holds col N = l%16; elements 0..7 are
// K = half*8 + 0..7, elements 8..15 are K = 16 + half*8 + 0..7. LDS B is stored [N][K].
__device__ __forceinline__ v16bf load_frag_b(const unsigned short* base, int col, int half) {
  BfPack p;
  p.h[0] = *reinterpret_cast<const v8us*>(base + col * KP + half * 8);
  p.h[1] = *reinterpret_cast<const v8us*>(base + col * KP + 16 + half * 8);
  return p.bf;
}

__global__ __launch_bounds__(256)
void linear_bf16x2_wmma(const float* __restrict__ X,
                        const float* __restrict__ W,
                        const float* __restrict__ bias,
                        float* __restrict__ Out) {
  // Double-buffered bf16 hi/lo planes: A as [M][K], B transposed to [N][K].
  // 4 arrays * 2 bufs * 5120 ushorts = 80 KB (WGP has 320 KB).
  __shared__ __align__(16) unsigned short As_hi[2][TILE_M * KP];
  __shared__ __align__(16) unsigned short As_lo[2][TILE_M * KP];
  __shared__ __align__(16) unsigned short Bs_hi[2][TILE_N * KP];
  __shared__ __align__(16) unsigned short Bs_lo[2][TILE_N * KP];

  const int tid  = threadIdx.x;
  const int lane = tid & 31;
  const int wave = tid >> 5;
  const int wm   = wave & 1;   // wave row (0..1): 64 rows each
  const int wn   = wave >> 1;  // wave col (0..3): 32 cols each
  const int half = lane >> 4;  // lane group 0/1
  const int lr   = lane & 15;

  const int mBase = blockIdx.y * TILE_M;
  const int nBase = blockIdx.x * TILE_N;

  // ---- per-thread global-load coordinates (float4 granularity) ----
  // A tile: 128 rows x 32 K floats = 1024 float4 -> 4 per thread
  int arow[4], akq[4];
#pragma unroll
  for (int i = 0; i < 4; ++i) {
    int idx = tid + i * 256;
    arow[i] = idx >> 3;   // 0..127
    akq[i]  = idx & 7;    // float4 slot in K (0..7)
  }
  // B tile: 32 K rows x 128 N floats = 1024 float4 -> 4 per thread
  int bkr[4], bnq[4];
#pragma unroll
  for (int i = 0; i < 4; ++i) {
    int idx = tid + i * 256;
    bkr[i] = idx >> 5;    // 0..31  (K row)
    bnq[i] = idx & 31;    // float4 slot in N (0..31)
  }

  float4 regA[4];
  float4 regB[4];

  auto load_tiles = [&](int kc) {
#pragma unroll
    for (int i = 0; i < 4; ++i) {
      const float* p = X + (size_t)(mBase + arow[i]) * K_DIM + kc * TILE_K + akq[i] * 4;
      regA[i] = *reinterpret_cast<const float4*>(p);
    }
#pragma unroll
    for (int i = 0; i < 4; ++i) {
      const float* p = W + (size_t)(kc * TILE_K + bkr[i]) * N_DIM + nBase + bnq[i] * 4;
      regB[i] = *reinterpret_cast<const float4*>(p);
    }
  };

  auto stage_to_lds = [&](int buf) {
    // A: K-contiguous -> pack 4 hi / 4 lo shorts and store as one b64 each.
#pragma unroll
    for (int i = 0; i < 4; ++i) {
      float v[4] = {regA[i].x, regA[i].y, regA[i].z, regA[i].w};
      unsigned short hi[4], lo[4];
#pragma unroll
      for (int e = 0; e < 4; ++e) bf_split(v[e], hi[e], lo[e]);
      v4us h4 = {hi[0], hi[1], hi[2], hi[3]};
      v4us l4 = {lo[0], lo[1], lo[2], lo[3]};
      int off = arow[i] * KP + akq[i] * 4;           // 8B-aligned
      *reinterpret_cast<v4us*>(&As_hi[buf][off]) = h4;
      *reinterpret_cast<v4us*>(&As_lo[buf][off]) = l4;
    }
    // B: transposed [N][K] -> b16 scatter; e*KP deltas fold into DS offsets.
#pragma unroll
    for (int i = 0; i < 4; ++i) {
      float v[4] = {regB[i].x, regB[i].y, regB[i].z, regB[i].w};
      unsigned short* bh = &Bs_hi[buf][bnq[i] * 4 * KP + bkr[i]];
      unsigned short* bl = &Bs_lo[buf][bnq[i] * 4 * KP + bkr[i]];
#pragma unroll
      for (int e = 0; e < 4; ++e) {
        unsigned short hi, lo;
        bf_split(v[e], hi, lo);
        bh[e * KP] = hi;
        bl[e * KP] = lo;
      }
    }
  };

  v8f acc[4][2] = {};  // 4 M-frags x 2 N-frags per wave (64x32 region)

  load_tiles(0);
  stage_to_lds(0);
  __syncthreads();

  for (int kc = 0; kc < NKC; ++kc) {
    const int  buf  = kc & 1;
    const bool more = (kc + 1) < NKC;
    if (more) load_tiles(kc + 1);              // overlap global latency w/ compute
    if (kc + 2 < NKC) {                        // gfx1250 global_prefetch_b8
      __builtin_prefetch(X + (size_t)(mBase + arow[0]) * K_DIM + (kc + 2) * TILE_K, 0, 1);
      __builtin_prefetch(W + (size_t)((kc + 2) * TILE_K + bkr[0]) * N_DIM + nBase, 0, 1);
    }

    v16bf aH[4], aL[4], bH[2], bL[2];
#pragma unroll
    for (int fm = 0; fm < 4; ++fm) {
      int row = wm * 64 + fm * 16 + lr;
      aH[fm] = load_frag_a(As_hi[buf], row, half);
      aL[fm] = load_frag_a(As_lo[buf], row, half);
    }
#pragma unroll
    for (int fn = 0; fn < 2; ++fn) {
      int col = wn * 32 + fn * 16 + lr;
      bH[fn] = load_frag_b(Bs_hi[buf], col, half);
      bL[fn] = load_frag_b(Bs_lo[buf], col, half);
    }

#pragma unroll
    for (int fm = 0; fm < 4; ++fm) {
#pragma unroll
      for (int fn = 0; fn < 2; ++fn) {
        v8f c = acc[fm][fn];
        // 2-term bf16 split of fp32: hi*hi + hi*lo + lo*hi (fp32 accumulate)
        c = __builtin_amdgcn_wmma_f32_16x16x32_bf16(false, aH[fm], false, bH[fn],
                                                    (short)0, c, false, false);
        c = __builtin_amdgcn_wmma_f32_16x16x32_bf16(false, aH[fm], false, bL[fn],
                                                    (short)0, c, false, false);
        c = __builtin_amdgcn_wmma_f32_16x16x32_bf16(false, aL[fm], false, bH[fn],
                                                    (short)0, c, false, false);
        acc[fm][fn] = c;
      }
    }

    // Stage next chunk into the *other* buffer while this one is being read;
    // a single barrier per iteration publishes it for the next chunk.
    if (more) stage_to_lds(buf ^ 1);
    __syncthreads();
  }

  // Epilogue: bias add + store. C/D layout: VGPR v -> row (v + 8*half), col lr.
#pragma unroll
  for (int fm = 0; fm < 4; ++fm) {
#pragma unroll
    for (int fn = 0; fn < 2; ++fn) {
      int col = nBase + wn * 32 + fn * 16 + lr;
      float bv = bias[col];
      int rbase = mBase + wm * 64 + fm * 16 + half * 8;
#pragma unroll
      for (int v = 0; v < 8; ++v) {
        Out[(size_t)(rbase + v) * N_DIM + col] = acc[fm][fn][v] + bv;
      }
    }
  }
}

extern "C" void kernel_launch(void* const* d_in, const int* in_sizes, int n_in,
                              void* d_out, int out_size, void* d_ws, size_t ws_size,
                              hipStream_t stream) {
  (void)in_sizes; (void)n_in; (void)out_size; (void)d_ws; (void)ws_size;
  const float* x    = (const float*)d_in[0];
  const float* w    = (const float*)d_in[1];
  const float* bias = (const float*)d_in[2];
  float* out        = (float*)d_out;

  dim3 grid(N_DIM / TILE_N, B_DIM / TILE_M);  // (32, 16)
  dim3 block(256);                             // 8 waves (wave32)
  linear_bf16x2_wmma<<<grid, block, 0, stream>>>(x, w, bias, out);
}
